// TripletLoss_9689446219925
// MI455X (gfx1250) — matrix operand order repaired and checked by reading, not compiled, not run on hardware
//
#include <hip/hip_runtime.h>

typedef __attribute__((ext_vector_type(16))) __bf16 bf16x16;
typedef __attribute__((ext_vector_type(8)))  __bf16 bf16x8;
typedef __attribute__((ext_vector_type(8)))  float  f32x8;
typedef __attribute__((ext_vector_type(4)))  float  f32x4;
typedef __attribute__((ext_vector_type(4)))  int    i32x4;

#define TL_MARGIN 0.3f
#define TILE_K 32
#define BM 128
#define BN 128
#define LDSS 40   // 32 + 8 pad -> 80-byte rows, keeps 16B chunk alignment, spreads banks

// ---- CDNA5 async global->LDS path (guarded; falls back to VGPR staging) ----
#if defined(__has_builtin)
#  if __has_builtin(__builtin_amdgcn_global_load_async_to_lds_b128)
#    define TL_ASYNC_LDS 1
#  endif
#endif
#ifndef TL_ASYNC_LDS
#  define TL_ASYNC_LDS 0
#endif

#if TL_ASYNC_LDS
#  if __has_builtin(__builtin_amdgcn_s_wait_asynccnt)
#    define TL_WAIT_ASYNC(n) __builtin_amdgcn_s_wait_asynccnt(n)
#  else
#    define TL_WAIT_ASYNC(n) asm volatile("s_wait_asynccnt %0" :: "n"(n) : "memory")
#  endif
#else
#  define TL_WAIT_ASYNC(n) ((void)0)
#endif

typedef __attribute__((address_space(1))) i32x4 tl_g_i32x4;  // global (AS1)
typedef __attribute__((address_space(3))) i32x4 tl_l_i32x4;  // LDS (AS3)

__device__ __forceinline__ void tl_copy_b128(const __bf16* g, __bf16* l) {
#if TL_ASYNC_LDS
    __builtin_amdgcn_global_load_async_to_lds_b128(
        (tl_g_i32x4*)(void*)const_cast<__bf16*>(g),
        (tl_l_i32x4*)(void*)l,
        0, 0);
#else
    *(bf16x8*)l = *(const bf16x8*)g;
#endif
}

// ---------------------------------------------------------------------------
// Kernel 1: f32 -> (bf16 hi, bf16 lo) split + exact f32 row squared-norms.
// ---------------------------------------------------------------------------
__global__ __launch_bounds__(256)
void tl_prep(const float* __restrict__ x, __bf16* __restrict__ a_hi,
             __bf16* __restrict__ a_lo, float* __restrict__ sq, int K) {
    const int row = blockIdx.x;
    const int t   = threadIdx.x;
    const float* xr = x + (size_t)row * K;
    float s = 0.f;
    for (int c0 = t * 8; c0 < K; c0 += 256 * 8) {
        f32x4 v0 = *(const f32x4*)(xr + c0);
        f32x4 v1 = *(const f32x4*)(xr + c0 + 4);
        bf16x8 hi, lo;
#pragma unroll
        for (int i = 0; i < 8; ++i) {
            float v = (i < 4) ? v0[i] : v1[i - 4];
            s += v * v;
            __bf16 h = (__bf16)v;
            hi[i] = h;
            lo[i] = (__bf16)(v - (float)h);
        }
        *(bf16x8*)(a_hi + (size_t)row * K + c0) = hi;
        *(bf16x8*)(a_lo + (size_t)row * K + c0) = lo;
    }
    __shared__ float red[256];
    red[t] = s;
    __syncthreads();
    for (int off = 128; off > 0; off >>= 1) {
        if (t < off) red[t] += red[t + off];
        __syncthreads();
    }
    if (t == 0) sq[row] = red[0];
}

// ---------------------------------------------------------------------------
// Kernel 2: init per-row reductions. ap init 0 is safe: all distances >= 0 and
// the diagonal (same class, dist ~1e-6) is always a positive pair.
// ---------------------------------------------------------------------------
__global__ void tl_init(float* __restrict__ ap, float* __restrict__ an, int half) {
    int i = blockIdx.x * blockDim.x + threadIdx.x;
    if (i < half) { ap[i] = 0.0f; an[i] = 1e30f; }
}

__device__ __forceinline__ bf16x16 tl_cat(bf16x8 a, bf16x8 b) {
    bf16x16 r;
#pragma unroll
    for (int i = 0; i < 8; ++i) { r[i] = a[i]; r[i + 8] = b[i]; }
    return r;
}

// ---------------------------------------------------------------------------
// Kernel 3: fused tiled GEMM (bf16 hi/lo x3 WMMA) + distance + masked row
// max/min. Double-buffered LDS fed by async global->LDS DMA (ASYNCcnt),
// overlapping the next tile's copy with the current tile's WMMAs.
// Block = 256 threads = 8 waves; wave grid 4x2; wave tile 32x64 (2x4 WMMA).
// ---------------------------------------------------------------------------
__global__ __launch_bounds__(256)
void tl_gemm(const __bf16* __restrict__ a_hi, const __bf16* __restrict__ a_lo,
             const float* __restrict__ sq, const int* __restrict__ tgt,
             float* __restrict__ ap_g, float* __restrict__ an_g, int K) {
    __shared__ __bf16 Ahi[2][BM * LDSS], Alo[2][BM * LDSS];
    __shared__ __bf16 Bhi[2][BN * LDSS], Blo[2][BN * LDSS];
    __shared__ int   trow[BM], tcol[BN];
    __shared__ float sqr[BM],  sqc[BN];

    const int tid    = threadIdx.x;
    const int lane   = tid & 31;
    const int wave   = tid >> 5;     // 0..7
    const int wr     = wave >> 1;    // 0..3 row group (32 rows each)
    const int wc     = wave & 1;     // 0..1 col group (64 cols each)
    const int halfid = lane >> 4;    // 0/1
    const int l15    = lane & 15;

    const int rowBase = blockIdx.x * BM;   // rows: first half only
    const int colBase = blockIdx.y * BN;   // cols: all n

    if (tid < BM)            { trow[tid] = tgt[rowBase + tid]; sqr[tid] = sq[rowBase + tid]; }
    else if (tid < BM + BN)  { int u = tid - BM; tcol[u] = tgt[colBase + u]; sqc[u] = sq[colBase + u]; }

    // Issue this thread's 8 async b128 copies for one 128x32 hi/lo A+B slab.
    auto issue_tile = [&](int buf, int kk) {
        for (int c = tid; c < (BM * TILE_K) / 8; c += 256) {
            const int r  = c >> 2;
            const int cc = (c & 3) * 8;
            const size_t ga = (size_t)(rowBase + r) * K + kk + cc;
            const size_t gb = (size_t)(colBase + r) * K + kk + cc;
            tl_copy_b128(a_hi + ga, &Ahi[buf][r * LDSS + cc]);
            tl_copy_b128(a_lo + ga, &Alo[buf][r * LDSS + cc]);
            tl_copy_b128(a_hi + gb, &Bhi[buf][r * LDSS + cc]);
            tl_copy_b128(a_lo + gb, &Blo[buf][r * LDSS + cc]);
        }
    };

    f32x8 acc[2][4];
#pragma unroll
    for (int mt = 0; mt < 2; ++mt)
#pragma unroll
        for (int nt = 0; nt < 4; ++nt) acc[mt][nt] = (f32x8){};

    const int nsteps = K / TILE_K;
    issue_tile(0, 0);                               // prologue: fill buffer 0

    for (int s = 0; s < nsteps; ++s) {
        const int  buf  = s & 1;
        const bool more = (s + 1) < nsteps;
        if (more) {
            issue_tile(buf ^ 1, (s + 1) * TILE_K);  // overlap next tile's DMA
            TL_WAIT_ASYNC(8);                       // oldest 8 (current buf) done
        } else {
            TL_WAIT_ASYNC(0);
        }
        __syncthreads();                            // all waves' tile data visible

        // A fragments: 16-bit A 16x32 lane layout (ISA 7.12.2).
        bf16x16 afh[2], afl[2];
#pragma unroll
        for (int mt = 0; mt < 2; ++mt) {
            const int r = wr * 32 + mt * 16 + l15;
            const int o = r * LDSS + halfid * 8;
            afh[mt] = tl_cat(*(const bf16x8*)&Ahi[buf][o], *(const bf16x8*)&Ahi[buf][o + 16]);
            afl[mt] = tl_cat(*(const bf16x8*)&Alo[buf][o], *(const bf16x8*)&Alo[buf][o + 16]);
        }
        // B fragments: 16-bit B 32x16, same K order as A (k = 8h+s, 16+8h+s-8).
        bf16x16 bfh[4], bfl[4];
#pragma unroll
        for (int nt = 0; nt < 4; ++nt) {
            const int r = wc * 64 + nt * 16 + l15;
            const int o = r * LDSS + halfid * 8;
            bfh[nt] = tl_cat(*(const bf16x8*)&Bhi[buf][o], *(const bf16x8*)&Bhi[buf][o + 16]);
            bfl[nt] = tl_cat(*(const bf16x8*)&Blo[buf][o], *(const bf16x8*)&Blo[buf][o + 16]);
        }

#pragma unroll
        for (int mt = 0; mt < 2; ++mt)
#pragma unroll
            for (int nt = 0; nt < 4; ++nt) {
                acc[mt][nt] = __builtin_amdgcn_wmma_f32_16x16x32_bf16(
                    false, afh[mt], false, bfh[nt], (short)0, acc[mt][nt], false, false);
                acc[mt][nt] = __builtin_amdgcn_wmma_f32_16x16x32_bf16(
                    false, afh[mt], false, bfl[nt], (short)0, acc[mt][nt], false, false);
                acc[mt][nt] = __builtin_amdgcn_wmma_f32_16x16x32_bf16(
                    false, afl[mt], false, bfh[nt], (short)0, acc[mt][nt], false, false);
            }
        __syncthreads();   // all reads of `buf` done before it is re-filled
    }

    // Epilogue: C/D layout -> element (m = r + 8*halfid, n = l15) per VGPR r.
#pragma unroll
    for (int mt = 0; mt < 2; ++mt) {
#pragma unroll
        for (int r = 0; r < 8; ++r) {
            const int lm = wr * 32 + mt * 16 + halfid * 8 + r;  // local row 0..127
            const float si = sqr[lm];
            const int   ti = trow[lm];
            float ap = 0.f, an = 1e30f;
#pragma unroll
            for (int nt = 0; nt < 4; ++nt) {
                const int ln = wc * 64 + nt * 16 + l15;
                const float S  = acc[mt][nt][r];
                const float d2 = si + sqc[ln] - 2.0f * S;
                const float dd = __builtin_sqrtf(fmaxf(d2, 1e-12f));
                const bool same = (ti == tcol[ln]);
                ap = same ? fmaxf(ap, dd) : ap;
                an = same ? an : fminf(an, dd);
            }
#pragma unroll
            for (int off = 1; off < 16; off <<= 1) {   // reduce within 16-lane half
                ap = fmaxf(ap, __shfl_xor(ap, off, 32));
                an = fminf(an, __shfl_xor(an, off, 32));
            }
            if (l15 == 0) {
                const int grow = rowBase + lm;
                atomicMax((int*)&ap_g[grow], __float_as_int(ap));  // all vals >= 0
                atomicMin((int*)&an_g[grow], __float_as_int(an));
            }
        }
    }
}

// ---------------------------------------------------------------------------
// Kernel 4: loss = mean(relu(ap - an + margin))
// ---------------------------------------------------------------------------
__global__ __launch_bounds__(256)
void tl_loss(const float* __restrict__ ap, const float* __restrict__ an,
             float* __restrict__ out, int half) {
    __shared__ float red[256];
    const int t = threadIdx.x;
    float s = 0.f;
    for (int i = t; i < half; i += 256) s += fmaxf(ap[i] - an[i] + TL_MARGIN, 0.f);
    red[t] = s;
    __syncthreads();
    for (int off = 128; off > 0; off >>= 1) {
        if (t < off) red[t] += red[t + off];
        __syncthreads();
    }
    if (t == 0) out[0] = red[0] / (float)half;
}

extern "C" void kernel_launch(void* const* d_in, const int* in_sizes, int n_in,
                              void* d_out, int out_size, void* d_ws, size_t ws_size,
                              hipStream_t stream) {
    const float* x   = (const float*)d_in[0];
    const int*   tgt = (const int*)d_in[1];
    const int n    = in_sizes[1];            // 4096
    const int K    = in_sizes[0] / n;        // 2048
    const int half = n / 2;

    const size_t nk = (size_t)n * (size_t)K;
    char* ws = (char*)d_ws;
    __bf16* a_hi = (__bf16*)ws;                       // nk * 2 bytes
    __bf16* a_lo = (__bf16*)(ws + nk * 2);            // nk * 2 bytes
    float*  sq   = (float*)(ws + nk * 4);             // n floats
    float*  ap   = sq + n;                            // half floats
    float*  an   = ap + half;                         // half floats
    float*  out  = (float*)d_out;

    tl_prep<<<n, 256, 0, stream>>>(x, a_hi, a_lo, sq, K);
    tl_init<<<(half + 255) / 256, 256, 0, stream>>>(ap, an, half);
    dim3 grid(half / BM, n / BN);
    tl_gemm<<<grid, 256, 0, stream>>>(a_hi, a_lo, sq, tgt, ap, an, K);
    tl_loss<<<1, 256, 0, stream>>>(ap, an, out, half);
}
